// ForwardDeformer_84963043049527
// MI455X (gfx1250) — matrix operand order repaired and checked by reading, not compiled, not run on hardware
//
#include <hip/hip_runtime.h>
#include <hip/hip_bf16.h>
#include <math.h>

// ---------------- problem constants (from reference) ----------------
#define RES        128
#define DD         64
#define J_BONES    24
#define G_VOX      (DD * RES * RES)      // 1,048,576 voxels
#define NPTS       131072
#define N_INIT     9
#define M_CAND     (NPTS * N_INIT)       // 1,179,648 candidates
#define NEWTON_IT  10
#define SOFTBLEND  20.0f
#define CVG_THR    1e-4f
#define PI_F       3.14159265358979323846f

typedef __attribute__((ext_vector_type(16))) _Float16 v16h;
typedef __attribute__((ext_vector_type(8)))  _Float16 v8h;
typedef __attribute__((ext_vector_type(8)))  float    v8f;

// wave-local LDS ordering: DS ops from one wave are in-order; this fence just
// stops the compiler from reordering and drains DScnt (gfx1250 split counter).
__device__ __forceinline__ void wave_lds_fence() {
  asm volatile("s_wait_dscnt 0x0" ::: "memory");
}

// 3x3 inverse via adjugate (v_rcp_f32 instead of IEEE divide)
__device__ __forceinline__ void inv3x3(const float r[9], float o[9]) {
  float c00 = r[4]*r[8] - r[5]*r[7];
  float c01 = r[5]*r[6] - r[3]*r[8];
  float c02 = r[3]*r[7] - r[4]*r[6];
  float det = r[0]*c00 + r[1]*c01 + r[2]*c02;
  float id  = __builtin_amdgcn_rcpf(det);
  o[0] = c00 * id;
  o[1] = (r[2]*r[7] - r[1]*r[8]) * id;
  o[2] = (r[1]*r[5] - r[2]*r[4]) * id;
  o[3] = c01 * id;
  o[4] = (r[0]*r[8] - r[2]*r[6]) * id;
  o[5] = (r[2]*r[3] - r[0]*r[5]) * id;
  o[6] = c02 * id;
  o[7] = (r[1]*r[6] - r[0]*r[7]) * id;
  o[8] = (r[0]*r[4] - r[1]*r[3]) * id;
}

// ---------------- kernel 0: affine inverses of the 9 init bones ----------------
__global__ void k_init_tinv(const float* __restrict__ tfs, float* __restrict__ tinv) {
  int i = threadIdx.x;
  if (i >= N_INIT) return;
  const int bones[N_INIT] = {0, 1, 2, 4, 5, 16, 17, 18, 19};
  int j = bones[i];
  float r[9], t[3];
  #pragma unroll
  for (int rr = 0; rr < 3; ++rr) {
    #pragma unroll
    for (int cc = 0; cc < 3; ++cc) r[rr*3 + cc] = tfs[j*16 + rr*4 + cc];
    t[rr] = tfs[j*16 + rr*4 + 3];
  }
  float ri[9];
  inv3x3(r, ri);
  #pragma unroll
  for (int q = 0; q < 9; ++q) tinv[i*12 + q] = ri[q];
  tinv[i*12 +  9] = -(ri[0]*t[0] + ri[1]*t[1] + ri[2]*t[2]);
  tinv[i*12 + 10] = -(ri[3]*t[0] + ri[4]*t[1] + ri[5]*t[2]);
  tinv[i*12 + 11] = -(ri[6]*t[0] + ri[7]*t[1] + ri[8]*t[2]);
}

// ---------------- kernel 1: WMMA-blended skinning field ----------------
// Per wave iteration: 32 voxels. Each lane owns one voxel's 24-bone softmax
// (no duplicated lbs reads). Weights staged in per-wave LDS laid out so each
// A fragment is two ds_load_b128 per lane. Two v_wmma_f32_16x16x32_f16 per
// iteration (voxels 0-15 / 16-31); D tiles transposed through LDS; every lane
// runs the displacement + adjugate-inverse epilogue and stores 12 interleaved
// channels (48B = three b128 stores).
__global__ void k_build_field(const float* __restrict__ tfs,
                              const float* __restrict__ lbs,
                              float* __restrict__ fld) {
  __shared__ _Float16 Th[32 * 16];        // B staging, K-major, rows >=24 zero
  __shared__ _Float16 Wh[8][32 * 32];     // per-wave weights [voxel][k], k padded to 32
  __shared__ float    Dt[8][2 * 256];     // per-wave two 16x16 D tiles

  const int tid  = threadIdx.x;
  const int lane = tid & 31;
  const int wv   = tid >> 5;
  const int m    = lane & 15;             // A row / B,C/D column within a tile
  const int hi   = lane >> 4;             // half-wave selector

  // Stage bone transforms (f16) once per block.
  for (int idx = tid; idx < 32 * 16; idx += blockDim.x) {
    int k = idx >> 4, n = idx & 15;
    Th[idx] = (k < J_BONES) ? (_Float16)tfs[k*16 + n] : (_Float16)0.0f;
  }
  __syncthreads();

  // B fragment (loop-invariant). this code's contiguous k order (same as A):
  // lanes 0-15 hold K=0..15, lanes 16-31 hold K=16..31; VGPR v -> K=2v,2v+1.
  v16h bf;
  #pragma unroll
  for (int v = 0; v < 8; ++v) {
    int kb = 16*hi + 2*v;
    bf[2*v]     = Th[kb * 16 + m];
    bf[2*v + 1] = Th[(kb + 1) * 16 + m];
  }

  const int nwaves  = gridDim.x * (blockDim.x >> 5);
  const int gw      = blockIdx.x * (blockDim.x >> 5) + wv;
  const int ngroups = G_VOX / 32;
  const int iters   = (ngroups + nwaves - 1) / nwaves;

  for (int it = 0; it < iters; ++it) {
    int  grp = gw + it * nwaves;
    bool act = grp < ngroups;             // wave-uniform -> EXEC all-1 at WMMA
    if (act) {
      int g = grp * 32 + lane;

      // prefetch next group's lbs line (gfx1250 global_prefetch_b8)
      {
        int gn = g + nwaves * 32;
        if (gn < G_VOX) __builtin_prefetch(lbs + gn, 0, 1);
      }

      // --- softmax over 24 bones, one voxel per lane ---
      float w[J_BONES];
      float mx = -1e30f;
      #pragma unroll
      for (int j = 0; j < J_BONES; ++j) {
        float val = SOFTBLEND * lbs[(size_t)j * G_VOX + g];
        w[j] = val;
        mx = fmaxf(mx, val);
      }
      float s = 0.0f;
      #pragma unroll
      for (int j = 0; j < J_BONES; ++j) { w[j] = __expf(w[j] - mx); s += w[j]; }
      float is = __builtin_amdgcn_rcpf(s);

      _Float16* wrow = &Wh[wv][lane * 32];
      #pragma unroll
      for (int j = 0; j < J_BONES; ++j) wrow[j] = (_Float16)(w[j] * is);
      #pragma unroll
      for (int j = J_BONES; j < 32; ++j) wrow[j] = (_Float16)0.0f;

      wave_lds_fence();

      // --- A fragments from LDS (two b128 per tile per lane) ---
      // A layout: af[0..15] = K 16hi..16hi+15.
      const v8h* r0 = (const v8h*)&Wh[wv][m * 32];            // tile 0: voxels 0-15
      const v8h* r1 = (const v8h*)&Wh[wv][(16 + m) * 32];     // tile 1: voxels 16-31
      v8h a0lo = r0[2 * hi], a0hi = r0[2 * hi + 1];
      v8h a1lo = r1[2 * hi], a1hi = r1[2 * hi + 1];
      v16h af0, af1;
      #pragma unroll
      for (int q = 0; q < 8; ++q) {
        af0[q] = a0lo[q]; af0[8 + q] = a0hi[q];
        af1[q] = a1lo[q]; af1[8 + q] = a1hi[q];
      }

      v8f acc0 = {}, acc1 = {};
      acc0 = __builtin_amdgcn_wmma_f32_16x16x32_f16(false, af0, false, bf,
                                                    (short)0, acc0, false, false);
      acc1 = __builtin_amdgcn_wmma_f32_16x16x32_f16(false, af1, false, bf,
                                                    (short)0, acc1, false, false);

      // C/D layout: VGPR v -> row (v + 8*hi), column m. Transpose via LDS.
      #pragma unroll
      for (int v = 0; v < 8; ++v) {
        Dt[wv][(v + 8*hi) * 16 + m]       = acc0[v];
        Dt[wv][256 + (v + 8*hi) * 16 + m] = acc1[v];
      }

      wave_lds_fence();

      // --- epilogue: every lane owns one voxel of the 32 ---
      float e[16];
      #pragma unroll
      for (int q = 0; q < 16; ++q) e[q] = Dt[wv][hi * 256 + m * 16 + q];

      int xi = g & (RES - 1);
      int yi = (g >> 7) & (RES - 1);
      int zi = g >> 14;
      float px = (float)xi * (2.0f / (RES - 1)) - 1.0f;
      float py = (float)yi * (2.0f / (RES - 1)) - 1.0f;
      float pz = (float)zi * (2.0f / (DD  - 1)) - 1.0f;

      float dx = e[0]*px + e[1]*py + e[2] *pz + e[3]  - px;
      float dy = e[4]*px + e[5]*py + e[6] *pz + e[7]  - py;
      float dz = e[8]*px + e[9]*py + e[10]*pz + e[11] - pz;

      float r[9] = {e[0], e[1], e[2], e[4], e[5], e[6], e[8], e[9], e[10]};
      float ri[9];
      inv3x3(r, ri);

      float4* o = (float4*)(fld + (size_t)g * 12);
      o[0] = make_float4(dx,    dy,    dz,    ri[0]);
      o[1] = make_float4(ri[1], ri[2], ri[3], ri[4]);
      o[2] = make_float4(ri[5], ri[6], ri[7], ri[8]);

      wave_lds_fence();   // protect Wh/Dt reuse next iteration
    }
  }
}

// ---------------- trilinear sample of the 12-channel AoS field ----------------
__device__ __forceinline__ void sample12(const float* __restrict__ fld,
                                         float x, float y, float z,
                                         float f[12]) {
  float xg = fminf(fmaxf((x + 1.0f) * 0.5f * (float)(RES - 1), 0.0f), (float)(RES - 1));
  float yg = fminf(fmaxf((y + 1.0f) * 0.5f * (float)(RES - 1), 0.0f), (float)(RES - 1));
  float zg = fminf(fmaxf((z + 1.0f) * 0.5f * (float)(DD  - 1), 0.0f), (float)(DD  - 1));
  int x0 = min((int)xg, RES - 2);
  int y0 = min((int)yg, RES - 2);
  int z0 = min((int)zg, DD  - 2);
  float fx = xg - (float)x0, fy = yg - (float)y0, fz = zg - (float)z0;
  float wx[2] = {1.0f - fx, fx};
  float wy[2] = {1.0f - fy, fy};
  float wz[2] = {1.0f - fz, fz};

  #pragma unroll
  for (int c = 0; c < 12; ++c) f[c] = 0.0f;

  #pragma unroll
  for (int dz = 0; dz < 2; ++dz)
    #pragma unroll
    for (int dy = 0; dy < 2; ++dy)
      #pragma unroll
      for (int dx = 0; dx < 2; ++dx) {
        int g = ((z0 + dz) * RES + (y0 + dy)) * RES + (x0 + dx);
        float w = wz[dz] * wy[dy] * wx[dx];
        const float4* p = (const float4*)(fld + (size_t)g * 12);
        float4 a = p[0], b = p[1], c4 = p[2];
        f[0] += w * a.x;  f[1] += w * a.y;  f[2]  += w * a.z;  f[3]  += w * a.w;
        f[4] += w * b.x;  f[5] += w * b.y;  f[6]  += w * b.z;  f[7]  += w * b.w;
        f[8] += w * c4.x; f[9] += w * c4.y; f[10] += w * c4.z; f[11] += w * c4.w;
      }
}

// ---------------- kernel 2: Newton correspondence search + posi_enc ----------------
__global__ void k_search(const float* __restrict__ xd,
                         const float* __restrict__ tinv,
                         const float* __restrict__ fld,
                         float* __restrict__ out) {
  int mIdx = blockIdx.x * blockDim.x + threadIdx.x;
  if (mIdx >= M_CAND) return;
  int n = mIdx / N_INIT;
  int i = mIdx - n * N_INIT;

  float p0 = xd[n*3 + 0], p1 = xd[n*3 + 1], p2 = xd[n*3 + 2];
  const float* ti = tinv + i * 12;
  float x = ti[0]*p0 + ti[1]*p1 + ti[2]*p2 + ti[9];
  float y = ti[3]*p0 + ti[4]*p1 + ti[5]*p2 + ti[10];
  float z = ti[6]*p0 + ti[7]*p1 + ti[8]*p2 + ti[11];

  float f[12];
  #pragma unroll 1
  for (int it = 0; it < NEWTON_IT; ++it) {
    sample12(fld, x, y, z, f);
    float e0 = x + f[0] - p0;
    float e1 = y + f[1] - p1;
    float e2 = z + f[2] - p2;
    x -= f[3]*e0 + f[4] *e1 + f[5] *e2;
    y -= f[6]*e0 + f[7] *e1 + f[8] *e2;
    z -= f[9]*e0 + f[10]*e1 + f[11]*e2;
  }
  sample12(fld, x, y, z, f);
  float e0 = x + f[0] - p0;
  float e1 = y + f[1] - p1;
  float e2 = z + f[2] - p2;
  float nrm2 = e0*e0 + e1*e1 + e2*e2;
  bool valid = (nrm2 < CVG_THR * CVG_THR) &&
               (fabsf(x) <= 1.0f) && (fabsf(y) <= 1.0f) && (fabsf(z) <= 1.0f);

  // x_enc: [x(3), sin(12), cos(12)]
  float* enc = out + (size_t)mIdx * 27;
  enc[0] = x; enc[1] = y; enc[2] = z;
  float cv[3] = {x, y, z};
  #pragma unroll
  for (int c = 0; c < 3; ++c)
    #pragma unroll
    for (int q = 0; q < 4; ++q) {
      float ang = cv[c] * PI_F * (float)(1 << q);
      enc[3  + c*4 + q] = __sinf(ang);
      enc[15 + c*4 + q] = __cosf(ang);
    }

  float* xo = out + (size_t)M_CAND * 27 + (size_t)mIdx * 3;
  xo[0] = x; xo[1] = y; xo[2] = z;
  out[(size_t)M_CAND * 30 + (size_t)mIdx] = valid ? 1.0f : 0.0f;
}

// ---------------- host launcher ----------------
extern "C" void kernel_launch(void* const* d_in, const int* in_sizes, int n_in,
                              void* d_out, int out_size, void* d_ws, size_t ws_size,
                              hipStream_t stream) {
  const float* xd  = (const float*)d_in[0];   // [N,3]
  const float* tfs = (const float*)d_in[1];   // [1,J,4,4]
  const float* lbs = (const float*)d_in[2];   // [1,J,DD,RES,RES]
  float* out = (float*)d_out;

  float* tinv = (float*)d_ws;                       // 9*12 floats
  float* fld  = (float*)((char*)d_ws + 256);        // 12 * G_VOX floats (~48 MB)

  k_init_tinv<<<1, 32, 0, stream>>>(tfs, tinv);
  k_build_field<<<256, 256, 0, stream>>>(tfs, lbs, fld);   // 2048 waves, 16 groups each
  k_search<<<(M_CAND + 255) / 256, 256, 0, stream>>>(xd, tinv, fld, out);
}
